// ImageSparseAttention_63565515980942
// MI455X (gfx1250) — matrix operand order repaired and checked by reading, not compiled, not run on hardware
//
#include <hip/hip_runtime.h>

// Problem constants (match reference)
constexpr int BB  = 8;
constexpr int IBc = 1024;
constexpr int Sc  = 4096;
constexpr int Dc  = 512;
constexpr int KTOP = Sc / 2 + 2;   // 2050

typedef __attribute__((ext_vector_type(16))) __bf16 v16bf;
typedef __attribute__((ext_vector_type(8)))  float  v8f;
typedef int v4i __attribute__((vector_size(16)));   // matches builtin param type

union Frag16 { v16bf v; unsigned int d[8]; };

#if __has_builtin(__builtin_amdgcn_global_load_async_to_lds_b128)
#define HAVE_ASYNC 1
#else
#define HAVE_ASYNC 0
#endif

#define AS1 __attribute__((address_space(1)))
#define AS3 __attribute__((address_space(3)))

// 16-byte global->LDS copy: async DMA engine when available, else plain copy.
__device__ __forceinline__ void copy16(const __bf16* __restrict__ src, __bf16* dst) {
#if HAVE_ASYNC
  __builtin_amdgcn_global_load_async_to_lds_b128(
      (AS1 v4i*)(unsigned long long)(const void*)src,
      (AS3 v4i*)(unsigned int)(unsigned long long)(const void*)dst, 0, 0);
#else
  *(uint4*)dst = *(const uint4*)src;
#endif
}

__device__ __forceinline__ void async_wait0() {
#if HAVE_ASYNC
#if __has_builtin(__builtin_amdgcn_s_wait_asynccnt)
  __builtin_amdgcn_s_wait_asynccnt(0);
#else
  asm volatile("s_wait_asynccnt 0x0" ::: "memory");
#endif
#endif
}

// ---------------------------------------------------------------------------
// bf16 WMMA GEMM (all-transposed-B form):  C = alpha * (A @ Bm^T) + bias
//   A:  (M,K) bf16 row-major     Bm: (N,K) bf16 row-major
//   BIASMODE: 0 none, 1 per-column (bias[N]), 2 per-row (bias[M])
//   OUTBF16:  write C as bf16 (feeds another GEMM) or fp32
// Macro tile 128x128, K-step 32, double-buffered LDS filled by async
// global->LDS B128 copies. 8 waves (4m x 2n), each 32x64 via 2x4
// v_wmma_f32_16x16x32_bf16 accumulators.
// ---------------------------------------------------------------------------
template <bool OUTBF16, int BIASMODE>
__global__ __launch_bounds__(256) void gemm_t(
    const __bf16* __restrict__ A, const __bf16* __restrict__ Bm,
    const float* __restrict__ bias, void* __restrict__ Cv,
    int M, int N, int K, float alpha, long sA, long sB, long sC) {
  constexpr int LDK = 40;  // LDS row stride in bf16 (80B: 16B-aligned chunks)
  __shared__ __bf16 lA[2][128 * LDK];
  __shared__ __bf16 lB[2][128 * LDK];

  A  += (long)blockIdx.z * sA;
  Bm += (long)blockIdx.z * sB;
  const long cbase = (long)blockIdx.z * sC;
  float*  Cf = (float*)Cv;
  __bf16* Ch = (__bf16*)Cv;

  const int m0   = blockIdx.y * 128;
  const int n0   = blockIdx.x * 128;
  const int tid  = threadIdx.x;
  const int lane = tid & 31;
  const int wave = tid >> 5;
  const int wm   = wave & 3;   // 32-row slab
  const int wn   = wave >> 2;  // 64-col slab
  const int lm   = lane & 15;
  const int kh   = lane >> 4;

  // staging: thread covers 16 contiguous bf16 (two B128 chunks) of one row
  const int srow = tid >> 1;
  const int scol = (tid & 1) * 16;
  const __bf16* gA = A + (long)(m0 + srow) * K + scol;
  const __bf16* gB = Bm + (long)(n0 + srow) * K + scol;
  const int aoff = srow * LDK + scol;

  v8f acc[2][4];
#pragma unroll
  for (int i = 0; i < 2; i++)
#pragma unroll
    for (int j = 0; j < 4; j++)
#pragma unroll
      for (int r = 0; r < 8; r++) acc[i][j][r] = 0.0f;

  const int ntiles = K >> 5;
  // prime buffer 0
  copy16(gA, &lA[0][aoff]); copy16(gA + 8, &lA[0][aoff + 8]);
  copy16(gB, &lB[0][aoff]); copy16(gB + 8, &lB[0][aoff + 8]);
  gA += 32; gB += 32;

  int buf = 0;
  for (int kt = 0; kt < ntiles; ++kt) {
    async_wait0();
    __syncthreads();
    if (kt + 1 < ntiles) {  // prefetch next tile into other buffer
      const int ob = buf ^ 1;
      copy16(gA, &lA[ob][aoff]); copy16(gA + 8, &lA[ob][aoff + 8]);
      copy16(gB, &lB[ob][aoff]); copy16(gB + 8, &lB[ob][aoff + 8]);
      gA += 32; gB += 32;
    }

    // fragments (ISA 7.12.2 16-bit layouts)
    Frag16 fa[2], fb[4];
#pragma unroll
    for (int i = 0; i < 2; i++) {
      const __bf16* p = &lA[buf][(wm * 32 + i * 16 + lm) * LDK + kh * 8];
      fa[i].d[0] = *(const unsigned int*)(p + 0);
      fa[i].d[1] = *(const unsigned int*)(p + 2);
      fa[i].d[2] = *(const unsigned int*)(p + 4);
      fa[i].d[3] = *(const unsigned int*)(p + 6);
      fa[i].d[4] = *(const unsigned int*)(p + 16);
      fa[i].d[5] = *(const unsigned int*)(p + 18);
      fa[i].d[6] = *(const unsigned int*)(p + 20);
      fa[i].d[7] = *(const unsigned int*)(p + 22);
    }
#pragma unroll
    for (int j = 0; j < 4; j++) {
      const __bf16* p = &lB[buf][(wn * 64 + j * 16 + lm) * LDK + kh * 8];
#pragma unroll
      for (int t = 0; t < 8; t++) fb[j].d[t] = *(const unsigned int*)(p + 2 * t + (t >= 4 ? 8 : 0));
    }
#pragma unroll
    for (int i = 0; i < 2; i++)
#pragma unroll
      for (int j = 0; j < 4; j++)
        acc[i][j] = __builtin_amdgcn_wmma_f32_16x16x32_bf16(
            false, fa[i].v, false, fb[j].v, (short)0, acc[i][j], false, false);
    buf ^= 1;
  }

  // epilogue: lane holds col lm, rows r + 8*kh per fragment
#pragma unroll
  for (int j = 0; j < 4; j++) {
    const int col = n0 + wn * 64 + j * 16 + lm;
    const float cb = (BIASMODE == 1) ? bias[col] : 0.0f;
#pragma unroll
    for (int i = 0; i < 2; i++) {
      const int mrow = m0 + wm * 32 + i * 16 + 8 * kh;
#pragma unroll
      for (int r = 0; r < 8; r++) {
        const float rb = (BIASMODE == 2) ? bias[mrow + r] : cb;
        const float val = alpha * acc[i][j][r] + rb;
        const long idx = cbase + (long)(mrow + r) * N + col;
        if (OUTBF16) Ch[idx] = (__bf16)val; else Cf[idx] = val;
      }
    }
  }
}

// ---------------------------------------------------------------------------
// fp32 -> bf16 bulk convert (4 elems/thread, n multiple of 4)
// ---------------------------------------------------------------------------
__global__ __launch_bounds__(256) void f32_to_bf16_kernel(
    const float* __restrict__ src, __bf16* __restrict__ dst, long n) {
  const long i = ((long)blockIdx.x * 256 + threadIdx.x) * 4;
  if (i + 3 < n) {
    const float4 f = *(const float4*)(src + i);
    dst[i + 0] = (__bf16)f.x; dst[i + 1] = (__bf16)f.y;
    dst[i + 2] = (__bf16)f.z; dst[i + 3] = (__bf16)f.w;
  }
}

// ---------------------------------------------------------------------------
// total[s] = sum_j Wb2s[s, j]
// ---------------------------------------------------------------------------
__global__ __launch_bounds__(256) void rowsum_kernel(const float* __restrict__ W,
                                                     float* __restrict__ out, int cols) {
  __shared__ float red[256];
  const int row = blockIdx.x, tid = threadIdx.x;
  const float* w = W + (long)row * cols;
  float s = 0.f;
  for (int c = tid; c < cols; c += 256) s += w[c];
  red[tid] = s; __syncthreads();
  for (int k = 128; k > 0; k >>= 1) { if (tid < k) red[tid] += red[tid + k]; __syncthreads(); }
  if (tid == 0) out[row] = red[0];
}

// ---------------------------------------------------------------------------
// aw[i,s] = (total[s] + (e-1)*(W[s,i-1]+W[s,i]+W[s,i+1]))/denom_i + bb2s[s]
// ---------------------------------------------------------------------------
__global__ __launch_bounds__(256) void build_aw_kernel(
    const float* __restrict__ W, const float* __restrict__ bb2s,
    const float* __restrict__ total, float* __restrict__ aw) {
  const int s = blockIdx.x * 256 + threadIdx.x;
  const int i = blockIdx.y;
  const float E = 2.718281828459045f;
  const int nb = (i == 0 || i == IBc - 1) ? 2 : 3;
  const float denom = (float)nb * E + (float)(IBc - nb);
  float band = W[(long)s * IBc + i];
  if (i > 0)       band += W[(long)s * IBc + i - 1];
  if (i < IBc - 1) band += W[(long)s * IBc + i + 1];
  aw[(long)i * Sc + s] = (total[s] + (E - 1.0f) * band) / denom + bb2s[s];
}

// ---------------------------------------------------------------------------
// Per-row exact top-K (2050/4096) radix select; ties from lowest index.
// Emits sparsified row directly in bf16 (A-operand of the next GEMM).
// ---------------------------------------------------------------------------
__global__ __launch_bounds__(256) void topk_row_kernel(const float* __restrict__ src_m,
                                                       __bf16* __restrict__ dst_m) {
  __shared__ unsigned int keys[Sc];
  __shared__ int hist[256];
  __shared__ unsigned int sh_prefix;
  __shared__ int sh_krem;
  const int row = blockIdx.x, tid = threadIdx.x;
  const float* src = src_m + (long)row * Sc;
  __bf16* dst = dst_m + (long)row * Sc;

  for (int c = tid; c < Sc; c += 256) {
    unsigned int b = __float_as_uint(src[c]);
    keys[c] = (b & 0x80000000u) ? ~b : (b | 0x80000000u);
  }
  unsigned int prefix = 0; int krem = KTOP;
  __syncthreads();

  for (int round = 0; round < 4; round++) {
    const int shift = 24 - round * 8;
    const unsigned int pmask = (round == 0) ? 0u : (0xFFFFFFFFu << (shift + 8));
    hist[tid] = 0;
    __syncthreads();
    for (int c = tid; c < Sc; c += 256) {
      unsigned int k = keys[c];
      if ((k & pmask) == prefix) atomicAdd(&hist[(k >> shift) & 255], 1);
    }
    __syncthreads();
    if (tid == 0) {
      int cum = 0, d = 255;
      for (; d > 0; d--) {
        if (cum + hist[d] >= krem) break;
        cum += hist[d];
      }
      sh_prefix = prefix | ((unsigned int)d << shift);
      sh_krem = krem - cum;
    }
    __syncthreads();
    prefix = sh_prefix; krem = sh_krem;
    __syncthreads();
  }

  const unsigned int T = prefix;
  for (int c = tid; c < Sc; c += 256) {
    const unsigned int k = keys[c];
    const unsigned int b = (k & 0x80000000u) ? (k & 0x7FFFFFFFu) : ~k;
    dst[c] = (k > T) ? (__bf16)__uint_as_float(b) : (__bf16)0.0f;
  }
  __syncthreads();
  if (tid == 0) {
    int take = krem;
    const unsigned int bt = (T & 0x80000000u) ? (T & 0x7FFFFFFFu) : ~T;
    const __bf16 tv = (__bf16)__uint_as_float(bt);
    for (int c = 0; c < Sc && take > 0; c++)
      if (keys[c] == T) { dst[c] = tv; take--; }
  }
}

// ---------------------------------------------------------------------------
// Row softmax: fp32 logits in, bf16 probabilities out
// ---------------------------------------------------------------------------
__global__ __launch_bounds__(256) void softmax_bf16_kernel(
    const float* __restrict__ X, __bf16* __restrict__ P, int cols) {
  __shared__ float red[256];
  const int row = blockIdx.x, tid = threadIdx.x;
  const float* x = X + (long)row * cols;
  __bf16* p = P + (long)row * cols;
  float mx = -3.402823466e38f;
  for (int c = tid; c < cols; c += 256) mx = fmaxf(mx, x[c]);
  red[tid] = mx; __syncthreads();
  for (int k = 128; k > 0; k >>= 1) { if (tid < k) red[tid] = fmaxf(red[tid], red[tid + k]); __syncthreads(); }
  mx = red[0]; __syncthreads();
  float sum = 0.f;
  for (int c = tid; c < cols; c += 256) sum += __expf(x[c] - mx);
  red[tid] = sum; __syncthreads();
  for (int k = 128; k > 0; k >>= 1) { if (tid < k) red[tid] += red[tid + k]; __syncthreads(); }
  const float inv = 1.0f / red[0];
  for (int c = tid; c < cols; c += 256) p[c] = (__bf16)(__expf(x[c] - mx) * inv);
}

// ---------------------------------------------------------------------------
extern "C" void kernel_launch(void* const* d_in, const int* in_sizes, int n_in,
                              void* d_out, int out_size, void* d_ws, size_t ws_size,
                              hipStream_t stream) {
  (void)in_sizes; (void)n_in; (void)out_size; (void)ws_size;
  const float* text  = (const float*)d_in[0];
  const float* image = (const float*)d_in[1];
  const float* Wq = (const float*)d_in[2]; const float* bq = (const float*)d_in[3];
  const float* Wk = (const float*)d_in[4]; const float* bk = (const float*)d_in[5];
  const float* Wv = (const float*)d_in[6]; const float* bv = (const float*)d_in[7];
  const float* Wb2s = (const float*)d_in[8]; const float* bb2s = (const float*)d_in[9];
  float* out = (float*)d_out;

  char* ws = (char*)d_ws;
  __bf16* tbf   = (__bf16*)ws; ws += (size_t)BB * Sc * Dc * 2;    // 32 MB
  __bf16* ibf   = (__bf16*)ws; ws += (size_t)BB * IBc * Dc * 2;   //  8 MB
  __bf16* wqbf  = (__bf16*)ws; ws += (size_t)Dc * Dc * 2;
  __bf16* wkbf  = (__bf16*)ws; ws += (size_t)Dc * Dc * 2;
  __bf16* wvbf  = (__bf16*)ws; ws += (size_t)Dc * Dc * 2;
  __bf16* wbbf  = (__bf16*)ws; ws += (size_t)Sc * IBc * 2;        //  8 MB
  __bf16* qbf   = (__bf16*)ws; ws += (size_t)BB * IBc * Dc * 2;   //  8 MB
  __bf16* kT    = (__bf16*)ws; ws += (size_t)BB * Dc * Sc * 2;    // 32 MB (B,D,S)
  __bf16* vT    = (__bf16*)ws; ws += (size_t)BB * Dc * Sc * 2;    // 32 MB
  float*  aw    = (float*)ws;  ws += (size_t)IBc * Sc * 4;        // 16 MB
  __bf16* spbf  = (__bf16*)ws; ws += (size_t)IBc * Sc * 2;        //  8 MB
  __bf16* out1  = (__bf16*)ws; ws += (size_t)BB * IBc * Dc * 2;   //  8 MB
  __bf16* attbf = (__bf16*)ws; ws += (size_t)BB * IBc * IBc * 2;  // 16 MB
  float*  logit = (float*)ws;  ws += (size_t)BB * IBc * Sc * 4;   // 128 MB
  __bf16* probs = (__bf16*)ws; ws += (size_t)BB * IBc * Sc * 2;   // 64 MB
  float*  total = (float*)ws;  ws += (size_t)Sc * 4;

  const dim3 blk(256);
  const float inv_sqrt_d = 0.04419417382415922f;  // 1/sqrt(512)

  // one-shot fp32 -> bf16 operand conversion (halves streamed bytes)
  f32_to_bf16_kernel<<<(BB * Sc * Dc) / 1024, blk, 0, stream>>>(text, tbf, (long)BB * Sc * Dc);
  f32_to_bf16_kernel<<<(BB * IBc * Dc) / 1024, blk, 0, stream>>>(image, ibf, (long)BB * IBc * Dc);
  f32_to_bf16_kernel<<<(Dc * Dc) / 1024, blk, 0, stream>>>(Wq, wqbf, (long)Dc * Dc);
  f32_to_bf16_kernel<<<(Dc * Dc) / 1024, blk, 0, stream>>>(Wk, wkbf, (long)Dc * Dc);
  f32_to_bf16_kernel<<<(Dc * Dc) / 1024, blk, 0, stream>>>(Wv, wvbf, (long)Dc * Dc);
  f32_to_bf16_kernel<<<(Sc * IBc) / 1024, blk, 0, stream>>>(Wb2s, wbbf, (long)Sc * IBc);

  // q = image @ Wq^T + bq                     (M=B*IB, N=D, K=D) -> bf16
  gemm_t<true, 1><<<dim3(Dc / 128, (BB * IBc) / 128, 1), blk, 0, stream>>>(
      ibf, wqbf, bq, qbf, BB * IBc, Dc, Dc, 1.0f, 0, 0, 0);
  // kT_b = Wk @ text_b^T + bk (row bias)      (M=D, N=S, K=D) -> bf16 (D,S)
  gemm_t<true, 2><<<dim3(Sc / 128, Dc / 128, BB), blk, 0, stream>>>(
      wkbf, tbf, bk, kT, Dc, Sc, Dc, 1.0f, 0, (long)Sc * Dc, (long)Dc * Sc);
  gemm_t<true, 2><<<dim3(Sc / 128, Dc / 128, BB), blk, 0, stream>>>(
      wvbf, tbf, bv, vT, Dc, Sc, Dc, 1.0f, 0, (long)Sc * Dc, (long)Dc * Sc);

  // analytic banded-softmax @ Wb2s^T, then exact top-k sparsify -> bf16
  rowsum_kernel<<<Sc, blk, 0, stream>>>(Wb2s, total, IBc);
  build_aw_kernel<<<dim3(Sc / 256, IBc), blk, 0, stream>>>(Wb2s, bb2s, total, aw);
  topk_row_kernel<<<IBc, blk, 0, stream>>>(aw, spbf);

  // out1_b = sparse @ k_b = sparse @ kT_b^T   (M=IB, N=D, K=S) -> bf16
  gemm_t<true, 0><<<dim3(Dc / 128, IBc / 128, BB), blk, 0, stream>>>(
      spbf, kT, nullptr, out1, IBc, Dc, Sc, 1.0f, 0, (long)Dc * Sc, (long)IBc * Dc);
  // att_b = q_b @ out1_b^T / sqrt(D)          (M=IB, N=IB, K=D) -> bf16
  gemm_t<true, 0><<<dim3(IBc / 128, IBc / 128, BB), blk, 0, stream>>>(
      qbf, out1, nullptr, attbf, IBc, IBc, Dc, inv_sqrt_d,
      (long)IBc * Dc, (long)IBc * Dc, (long)IBc * IBc);
  // logit_b = att_b @ Wb2s^T + bb2s           (M=IB, N=S, K=IB) -> fp32
  gemm_t<false, 1><<<dim3(Sc / 128, IBc / 128, BB), blk, 0, stream>>>(
      attbf, wbbf, bb2s, logit, IBc, Sc, IBc, 1.0f,
      (long)IBc * IBc, 0, (long)IBc * Sc);
  // softmax rows -> bf16 probs
  softmax_bf16_kernel<<<BB * IBc, blk, 0, stream>>>(logit, probs, Sc);
  // out_b = probs_b @ v_b = probs_b @ vT_b^T  (M=IB, N=D, K=S) -> fp32
  gemm_t<false, 0><<<dim3(Dc / 128, IBc / 128, BB), blk, 0, stream>>>(
      probs, vT, nullptr, out, IBc, Dc, Sc, 1.0f,
      (long)IBc * Sc, (long)Dc * Sc, (long)IBc * Dc);
}
